// DynamicsFunction_13151189860706
// MI455X (gfx1250) — matrix operand order repaired and checked by reading, not compiled, not run on hardware
//
#include <hip/hip_runtime.h>

typedef _Float16 v16h __attribute__((ext_vector_type(16)));
typedef _Float16 h8   __attribute__((ext_vector_type(8)));
typedef float    v8f  __attribute__((ext_vector_type(8)));

#define NSEG  63
#define IN_D  32
#define HID_D 128
#define OUT_D 32

// ---------- helpers -------------------------------------------------------

__device__ __forceinline__ float fast_tanh(float x) {
#if __has_builtin(__builtin_amdgcn_tanhf)
    return __builtin_amdgcn_tanhf(x);
#elif __has_builtin(__builtin_amdgcn_tanh_f32)
    return __builtin_amdgcn_tanh_f32(x);
#else
    return tanhf(x);
#endif
}

__device__ __forceinline__ v8f splat8(float x) {
    v8f v;
#pragma unroll
    for (int r = 0; r < 8; ++r) v[r] = x;
    return v;
}

// Pre-swizzled B-matrix fragment (KxN f16 for v_wmma_f32_16x16x32_f16):
// element e of lane's v16h = W[kt*32 + (lane>>4)*8 + (e&7) + (e>>3)*16][nt*16 + (lane&15)]
__device__ __forceinline__ const v16h* bfrag(const _Float16* __restrict__ base,
                                             int kt_count, int nt, int kt, int lane) {
    return (const v16h*)(base + (((size_t)(nt * kt_count + kt) * 32 + lane) << 4));
}

// A fragment from row-major f16 LDS tile: two contiguous ds_load_b128.
__device__ __forceinline__ v16h load_afrag(const _Float16* lds, int ld,
                                           int kbase, int lane) {
    int m = lane & 15;
    int half = lane >> 4;
    const _Float16* p = lds + m * ld + kbase + half * 8;
    h8 lo = *(const h8*)p;
    h8 hi = *(const h8*)(p + 16);
    return __builtin_shufflevector(lo, hi, 0, 1, 2, 3, 4, 5, 6, 7,
                                   8, 9, 10, 11, 12, 13, 14, 15);
}

__device__ __forceinline__ v8f wmma_f16(v16h a, v16h b, v8f c) {
    return __builtin_amdgcn_wmma_f32_16x16x32_f16(false, a, false, b,
                                                  (short)0, c, false, false);
}

// ---------- weight prep: fp32 -> f16 B-fragment layout --------------------

__device__ __forceinline__ void fill_bfrag(_Float16* __restrict__ dst,
                                           const float* __restrict__ W,
                                           int K, int N, int rel) {
    int KT   = K >> 5;
    int e    = rel & 15;
    int lane = (rel >> 4) & 31;
    int kt   = (rel >> 9) % KT;
    int nt   = rel / (KT << 9);
    int n = nt * 16 + (lane & 15);
    int k = kt * 32 + (lane >> 4) * 8 + (e & 7) + (e >> 3) * 16;
    dst[rel] = (_Float16)W[(size_t)k * N + n];
}

#define W1F_OFF  0
#define W2F_OFF  16384      // 4*8*512
#define WIF_OFF  540672     // + 4*256*512
#define WROF_OFF 544768     // + 1*8*512
#define WE1F_OFF 548864     // + 4*2*512
#define WE2F_OFF 552960     // + 1*8*512
#define WF_TOTAL 557056     // + 4*2*512

__global__ void prep_weights(const float* __restrict__ Wi, const float* __restrict__ W1,
                             const float* __restrict__ W2, const float* __restrict__ Wro,
                             const float* __restrict__ We1, const float* __restrict__ We2,
                             _Float16* __restrict__ ws) {
    size_t gid = (size_t)blockIdx.x * blockDim.x + threadIdx.x;
    if (gid < W2F_OFF)        fill_bfrag(ws + W1F_OFF,  W1,  128, 128,  (int)gid);
    else if (gid < WIF_OFF)   fill_bfrag(ws + W2F_OFF,  W2,  128, 4096, (int)(gid - W2F_OFF));
    else if (gid < WROF_OFF)  fill_bfrag(ws + WIF_OFF,  Wi,  32,  128,  (int)(gid - WIF_OFF));
    else if (gid < WE1F_OFF)  fill_bfrag(ws + WROF_OFF, Wro, 128, 32,   (int)(gid - WROF_OFF));
    else if (gid < WE2F_OFF)  fill_bfrag(ws + WE1F_OFF, We1, 32,  128,  (int)(gid - WE1F_OFF));
    else if (gid < WF_TOTAL)  fill_bfrag(ws + WE2F_OFF, We2, 128, 32,   (int)(gid - WE2F_OFF));
}

// ---------- CDE vector field: k = einsum(tanh(relu(z@W1+b1)@W2+b2), dX) ---

__device__ __forceinline__ void cde_field_eval(
        const _Float16 (&zeval)[16 * HID_D], const float (&dx)[16 * IN_D],
        float (&kout)[16 * HID_D], _Float16 (&Hl)[16 * HID_D],
        const float (&bias2s)[4096],
        const _Float16* __restrict__ W1f, const _Float16* __restrict__ W2f,
        float bias1, int wave, int lane) {
    int n = lane & 15, half = lane >> 4;

    // dX values this lane needs: invariant over the whole phase-B sweep.
    float dxv[2][8];
#pragma unroll
    for (int p = 0; p < 2; ++p)
#pragma unroll
        for (int r = 0; r < 8; ++r)
            dxv[p][r] = dx[(r + 8 * half) * IN_D + ((p << 4) | n)];

    // Phase A: H = relu(zeval @ W1 + b1); bias folded into the C operand.
    {
        v8f c = splat8(bias1);
#pragma unroll
        for (int kt = 0; kt < 4; ++kt)
            c = wmma_f16(load_afrag(&zeval[0], HID_D, kt * 32, lane),
                         *bfrag(W1f, 4, wave, kt, lane), c);
        int col = wave * 16 + n;
#pragma unroll
        for (int r = 0; r < 8; ++r)
            Hl[(r + 8 * half) * HID_D + col] = (_Float16)(c[r] > 0.f ? c[r] : 0.f);
    }
    __syncthreads();

    // Phase B: G = tanh(H@W2+b2); contract over i with dX.
    // Wave w owns h = w*16..w*16+15 (2 N-tiles each of the 4096-wide output).
    v16h af[4];
#pragma unroll
    for (int kt = 0; kt < 4; ++kt) af[kt] = load_afrag(&Hl[0], HID_D, kt * 32, lane);

    // Linear fragment walk: wave's slab starts at nt = wave*32; each fragment
    // is 512 halves; per (nt) step is 4 fragments (kt).
    const _Float16* __restrict__ wp = W2f + ((size_t)wave * 32 * 4 * 512) + (lane << 4);
    int jbase = wave * 512 + n;   // bias/LDS column for (hh=0, p=0)

#pragma unroll 1
    for (int hh = 0; hh < 16; ++hh) {
        int h = wave * 16 + hh;
        float sum[8];
#pragma unroll
        for (int r = 0; r < 8; ++r) sum[r] = 0.f;
#pragma unroll
        for (int p = 0; p < 2; ++p) {
            v8f cc = splat8(bias2s[jbase + (hh * 2 + p) * 16]);
#pragma unroll
            for (int kt = 0; kt < 4; ++kt)
                cc = wmma_f16(af[kt], *(const v16h*)(wp + kt * 512), cc);
            wp += 4 * 512;
#pragma unroll
            for (int r = 0; r < 8; ++r)
                sum[r] += fast_tanh(cc[r]) * dxv[p][r];
        }
        // reduce over the 16 lanes of each half (same h, same m per VGPR r)
#pragma unroll
        for (int off = 1; off < 16; off <<= 1)
#pragma unroll
            for (int r = 0; r < 8; ++r) sum[r] += __shfl_xor(sum[r], off, 32);
        if (n == 0)
#pragma unroll
            for (int r = 0; r < 8; ++r) kout[(r + 8 * half) * HID_D + h] = sum[r];
    }
    __syncthreads();
}

// ---------- evolution field: k = tanh(y@We1+be1)@We2+be2 ------------------

__device__ __forceinline__ void ev_field_eval(
        const _Float16 (&yeval)[16 * OUT_D], float (&kout)[16 * OUT_D],
        _Float16 (&Hl)[16 * HID_D],
        const _Float16* __restrict__ We1f, const _Float16* __restrict__ We2f,
        const float* __restrict__ be1, const float* __restrict__ be2,
        int wave, int lane) {
    int n = lane & 15, half = lane >> 4;
    {   // Ht = tanh(y @ We1 + be1): K=32 (1 k-tile), N=128 (wave w -> tile w)
        v8f c = splat8(be1[wave * 16 + n]);
        c = wmma_f16(load_afrag(&yeval[0], IN_D, 0, lane),
                     *bfrag(We1f, 1, wave, 0, lane), c);
        int col = wave * 16 + n;
#pragma unroll
        for (int r = 0; r < 8; ++r)
            Hl[(r + 8 * half) * HID_D + col] = (_Float16)fast_tanh(c[r]);
    }
    __syncthreads();
    if (wave < 2) {  // Ht @ We2 + be2 : N=32 -> 2 tiles
        v8f c = splat8(be2[wave * 16 + n]);
#pragma unroll
        for (int kt = 0; kt < 4; ++kt)
            c = wmma_f16(load_afrag(&Hl[0], HID_D, kt * 32, lane),
                         *bfrag(We2f, 4, wave, kt, lane), c);
        int col = wave * 16 + n;
#pragma unroll
        for (int r = 0; r < 8; ++r)
            kout[(r + 8 * half) * OUT_D + col] = c[r];
    }
    __syncthreads();
}

// ---------- main persistent kernel: one 16-row batch tile per block -------

__global__ __launch_bounds__(256) void cde_main(
        const float* __restrict__ coeffs,
        const float* __restrict__ b_init, const float* __restrict__ b1,
        const float* __restrict__ b2, const float* __restrict__ b_ro,
        const float* __restrict__ be1, const float* __restrict__ be2,
        const _Float16* __restrict__ Wif, const _Float16* __restrict__ W1f,
        const _Float16* __restrict__ W2f, const _Float16* __restrict__ Wrof,
        const _Float16* __restrict__ We1f, const _Float16* __restrict__ We2f,
        float* __restrict__ out) {
    __shared__ float    zcur[16 * HID_D];
    __shared__ float    acc [16 * HID_D];
    __shared__ float    kcur[16 * HID_D];
    __shared__ _Float16 zeval[16 * HID_D];
    __shared__ _Float16 Hl  [16 * HID_D];
    __shared__ float    bias2s[4096];
    __shared__ float    dx1[16 * IN_D], dx2[16 * IN_D], dx4[16 * IN_D];
    __shared__ float    ycur[16 * OUT_D], yk[16 * OUT_D], yac[16 * OUT_D];
    __shared__ _Float16 yeval[16 * OUT_D];

    int tid = threadIdx.x;
    int lane = tid & 31;
    int wave = tid >> 5;
    int n = lane & 15, half = lane >> 4;
    int rowbase = blockIdx.x * 16;

    float bias1 = b1[wave * 16 + n];

    // Stage b2 into LDS once; reused by all 252 evals.
    for (int idx = tid; idx < 4096; idx += 256) bias2s[idx] = b2[idx];

    // ---- init: z0 = X0 @ W_init + b_init, X0 = coeffs[:,0,0:32] ----
    for (int idx = tid; idx < 16 * IN_D; idx += 256) {
        int m = idx >> 5, i = idx & 31;
        yeval[idx] = (_Float16)coeffs[((size_t)(rowbase + m) * NSEG) * (4 * IN_D) + i];
    }
    __syncthreads();
    {
        v8f c = splat8(b_init[wave * 16 + n]);
        c = wmma_f16(load_afrag(&yeval[0], IN_D, 0, lane),
                     *bfrag(Wif, 1, wave, 0, lane), c);
        int col = wave * 16 + n;
#pragma unroll
        for (int r = 0; r < 8; ++r) {
            int m = r + 8 * half;
            zcur[m * HID_D + col] = c[r];
            zeval[m * HID_D + col] = (_Float16)c[r];
        }
    }
    __syncthreads();

    // ---- 63 RK4 steps of the CDE ----
    for (int t = 0; t < NSEG; ++t) {
        for (int idx = tid; idx < 16 * IN_D; idx += 256) {
            int m = idx >> 5, i = idx & 31;
            size_t base = ((size_t)(rowbase + m) * NSEG + t) * (4 * IN_D);
            float cb = coeffs[base + IN_D + i];
            float cc = coeffs[base + 2 * IN_D + i];
            float cd = coeffs[base + 3 * IN_D + i];
            dx1[idx] = cb;                               // fr = 0
            dx2[idx] = cb + cc + 0.75f * cd;             // fr = 0.5
            dx4[idx] = (t < NSEG - 1)
                ? coeffs[base + 4 * IN_D + IN_D + i]     // next seg, fr = 0
                : (cb + 2.f * cc + 3.f * cd);            // clipped seg, fr = 1
        }
        __syncthreads();

        cde_field_eval(zeval, dx1, kcur, Hl, bias2s, W1f, W2f, bias1, wave, lane);
        for (int idx = tid; idx < 16 * HID_D; idx += 256) {
            float k = kcur[idx];
            acc[idx] = k;
            zeval[idx] = (_Float16)(zcur[idx] + 0.5f * k);
        }
        __syncthreads();
        cde_field_eval(zeval, dx2, kcur, Hl, bias2s, W1f, W2f, bias1, wave, lane);
        for (int idx = tid; idx < 16 * HID_D; idx += 256) {
            float k = kcur[idx];
            acc[idx] += 2.f * k;
            zeval[idx] = (_Float16)(zcur[idx] + 0.5f * k);
        }
        __syncthreads();
        cde_field_eval(zeval, dx2, kcur, Hl, bias2s, W1f, W2f, bias1, wave, lane);
        for (int idx = tid; idx < 16 * HID_D; idx += 256) {
            float k = kcur[idx];
            acc[idx] += 2.f * k;
            zeval[idx] = (_Float16)(zcur[idx] + k);
        }
        __syncthreads();
        cde_field_eval(zeval, dx4, kcur, Hl, bias2s, W1f, W2f, bias1, wave, lane);
        for (int idx = tid; idx < 16 * HID_D; idx += 256) {
            float z = zcur[idx] + (acc[idx] + kcur[idx]) * (1.f / 6.f);
            zcur[idx] = z;
            zeval[idx] = (_Float16)z;
        }
        __syncthreads();
    }

    // ---- pred_y = zT @ W_ro + b_ro ----
    if (wave < 2) {
        v8f c = splat8(b_ro[wave * 16 + n]);
#pragma unroll
        for (int kt = 0; kt < 4; ++kt)
            c = wmma_f16(load_afrag(&zeval[0], HID_D, kt * 32, lane),
                         *bfrag(Wrof, 4, wave, kt, lane), c);
        int col = wave * 16 + n;
#pragma unroll
        for (int r = 0; r < 8; ++r) {
            int m = r + 8 * half;
            ycur[m * OUT_D + col] = c[r];
            yeval[m * OUT_D + col] = (_Float16)c[r];
        }
    }
    __syncthreads();

    // ---- 10 RK4 steps of the evolution ODE (h = 0.5) ----
    const float hs = 0.5f;
    for (int s = 0; s < 10; ++s) {
        ev_field_eval(yeval, yk, Hl, We1f, We2f, be1, be2, wave, lane);
        for (int idx = tid; idx < 16 * OUT_D; idx += 256) {
            float k = yk[idx];
            yac[idx] = k;
            yeval[idx] = (_Float16)(ycur[idx] + 0.5f * hs * k);
        }
        __syncthreads();
        ev_field_eval(yeval, yk, Hl, We1f, We2f, be1, be2, wave, lane);
        for (int idx = tid; idx < 16 * OUT_D; idx += 256) {
            float k = yk[idx];
            yac[idx] += 2.f * k;
            yeval[idx] = (_Float16)(ycur[idx] + 0.5f * hs * k);
        }
        __syncthreads();
        ev_field_eval(yeval, yk, Hl, We1f, We2f, be1, be2, wave, lane);
        for (int idx = tid; idx < 16 * OUT_D; idx += 256) {
            float k = yk[idx];
            yac[idx] += 2.f * k;
            yeval[idx] = (_Float16)(ycur[idx] + hs * k);
        }
        __syncthreads();
        ev_field_eval(yeval, yk, Hl, We1f, We2f, be1, be2, wave, lane);
        for (int idx = tid; idx < 16 * OUT_D; idx += 256) {
            float y = ycur[idx] + (hs / 6.f) * (yac[idx] + yk[idx]);
            ycur[idx] = y;
            yeval[idx] = (_Float16)y;
        }
        __syncthreads();
    }

    for (int idx = tid; idx < 16 * OUT_D; idx += 256) {
        int m = idx >> 5, i = idx & 31;
        out[(size_t)(rowbase + m) * OUT_D + i] = ycur[idx];
    }
}

// ---------- launch --------------------------------------------------------

extern "C" void kernel_launch(void* const* d_in, const int* in_sizes, int n_in,
                              void* d_out, int out_size, void* d_ws, size_t ws_size,
                              hipStream_t stream) {
    const float* coeffs = (const float*)d_in[0];
    const float* W_init = (const float*)d_in[1];
    const float* b_init = (const float*)d_in[2];
    const float* W1     = (const float*)d_in[3];
    const float* b1     = (const float*)d_in[4];
    const float* W2     = (const float*)d_in[5];
    const float* b2     = (const float*)d_in[6];
    const float* W_ro   = (const float*)d_in[7];
    const float* b_ro   = (const float*)d_in[8];
    const float* We1    = (const float*)d_in[9];
    const float* be1    = (const float*)d_in[10];
    const float* We2    = (const float*)d_in[11];
    const float* be2    = (const float*)d_in[12];
    float* out = (float*)d_out;
    _Float16* wsf = (_Float16*)d_ws;

    prep_weights<<<(WF_TOTAL + 255) / 256, 256, 0, stream>>>(
        W_init, W1, W2, W_ro, We1, We2, wsf);

    cde_main<<<2048 / 16, 256, 0, stream>>>(
        coeffs, b_init, b1, b2, b_ro, be1, be2,
        wsf + WIF_OFF, wsf + W1F_OFF, wsf + W2F_OFF,
        wsf + WROF_OFF, wsf + WE1F_OFF, wsf + WE2F_OFF,
        out);
}
